// Vector_Quantizer_Block_5970004541982
// MI455X (gfx1250) — matrix operand order repaired and checked by reading, not compiled, not run on hardware
//
#include <hip/hip_runtime.h>
#include <hip/hip_bf16.h>

typedef __attribute__((ext_vector_type(16))) __bf16 v16bf;
typedef __attribute__((ext_vector_type(8)))  float  v8f;

#define EC       256                 // embedding dim (C)
#define NK       512                 // number of codes
#define KC       128                 // codes staged per LDS chunk
#define NCHUNK   (NK / KC)           // 4
#define ESTRIDE  136                 // padded code stride in LDS (68 dwords -> 2-way conflicts max)
#define HW       1024                // 32*32 spatial positions per batch image
#define ROWS_PER_WG 128              // 8 waves * 16 rows
#define NROWS    65536               // 64*32*32
#define NWG      (NROWS / ROWS_PER_WG) // 512
#define Q_SIZE   (64 * 256 * 32 * 32)  // 16777216
#define INV_TOTAL (1.0f / 16777216.0f) // 1 / (N*C)

// LDS layout (dynamic):
//   elds   : bf16 [EC][ESTRIDE]   transposed codebook chunk   69632 B
//   eNorm  : f32  [KC]            per-code squared norms        512 B
//   idxLDS : i32  [8*16]          per-row argmin result         512 B
//   wpart  : f32  [8]             per-wave loss partials         32 B
#define SMEM_ELDS   0
#define SMEM_ENORM  (EC * ESTRIDE * 2)
#define SMEM_IDX    (SMEM_ENORM + KC * 4)
#define SMEM_WPART  (SMEM_IDX + 8 * 16 * 4)
#define SMEM_BYTES  (SMEM_WPART + 8 * 4)

__global__ __launch_bounds__(256) void vq_wmma_kernel(
    const float* __restrict__ x, const float* __restrict__ emb,
    float* __restrict__ out, float* __restrict__ wsPartial) {
  extern __shared__ char smem[];
  __bf16* elds  = (__bf16*)(smem + SMEM_ELDS);
  float*  eNorm = (float*)(smem + SMEM_ENORM);
  int*    idxLDS = (int*)(smem + SMEM_IDX);
  float*  wpart = (float*)(smem + SMEM_WPART);

  const int tid  = threadIdx.x;
  const int wave = tid >> 5;
  const int lane = tid & 31;
  const int half = lane >> 4;   // which 16-lane half
  const int l16  = lane & 15;

  // ---- Rows owned by this wave: 16 consecutive spatial positions ----
  const int rowBase = blockIdx.x * ROWS_PER_WG + wave * 16;
  const int R    = rowBase + l16;        // this lane's A-matrix row
  const int bIdx = R >> 10;              // batch index
  const int hw   = R & (HW - 1);         // h*32+w
  const float* xrow = x + (((size_t)bIdx * EC) << 10) + hw;  // + (c<<10) per channel

  // ---- Load A fragments (16 rows x 256 dims) as bf16, resident in VGPRs ----
  // 16-bit A 16x32 layout: lanes 0-15 hold K {0..7,16..23}, lanes 16-31 hold {8..15,24..31}
  v16bf a[8];
#pragma unroll
  for (int kb = 0; kb < 8; ++kb) {
#pragma unroll
    for (int e = 0; e < 16; ++e) {
      const int c = kb * 32 + (e & 7) + ((e >> 3) << 4) + half * 8;
      a[kb][e] = (__bf16)xrow[(size_t)c << 10];
    }
  }

  float bestVal[8];
  int   bestIdx[8];
#pragma unroll
  for (int r = 0; r < 8; ++r) { bestVal[r] = 3.4e38f; bestIdx[r] = 0; }

  // ---- Sweep the codebook in LDS chunks of 128 codes ----
  for (int ch = 0; ch < NCHUNK; ++ch) {
    const int nBase = ch * KC;
    __syncthreads();  // previous chunk's readers done before restage

    // Stage: thread t owns dim column c=t; reads emb row-major (coalesced across
    // lanes), writes transposed bf16 (272B lane stride -> 2-way bank conflict max).
    {
      const int c = tid;
      for (int i = 0; i < KC; ++i) {
        elds[c * ESTRIDE + i] = (__bf16)emb[(size_t)(nBase + i) * EC + c];
      }
    }
    __syncthreads();

    // Per-code squared norm from the same bf16 values used in the WMMA dot.
    if (tid < KC) {
      float s = 0.f;
      for (int c2 = 0; c2 < EC; ++c2) {
        float v = (float)elds[c2 * ESTRIDE + tid];
        s += v * v;
      }
      eNorm[tid] = s;
    }
    __syncthreads();

    // Prefetch next chunk of the codebook into cache (global_prefetch_b8).
    if (ch + 1 < NCHUNK) {
      __builtin_prefetch(&emb[(size_t)(nBase + KC) * EC + (size_t)tid * 128], 0, 0);
    }

    // Compute: 8 N-tiles of 16 codes; K=256 -> 8 WMMA steps per tile.
    for (int nt = 0; nt < KC / 16; ++nt) {
      const int n0 = nt * 16;
      v8f acc = {};
#pragma unroll
      for (int kb = 0; kb < 8; ++kb) {
        // B 32x16 layout: lane holds column n = n0 + l%16; K in same table order as A.
        union { uint4 u[2]; v16bf v; } bf;
#pragma unroll
        for (int e = 0; e < 16; ++e) {
          const int Kr = kb * 32 + (e & 7) + ((e >> 3) << 4) + half * 8;
          bf.v[e] = elds[Kr * ESTRIDE + n0 + l16];
        }
        acc = __builtin_amdgcn_wmma_f32_16x16x32_bf16(
            false, a[kb], false, bf.v, (short)0, acc, false, false);
      }
      // d' = ||e||^2 - 2 x.e  (||x||^2 constant per row, irrelevant for argmin)
      const float en   = eNorm[n0 + l16];
      const int   nIdx = nBase + n0 + l16;
#pragma unroll
      for (int r = 0; r < 8; ++r) {
        const float d = en - 2.0f * acc[r];
        if (d < bestVal[r]) { bestVal[r] = d; bestIdx[r] = nIdx; }
      }
    }
  }

  // ---- Argmin across the 16 lanes of each half (C/D layout: lane = column) ----
#pragma unroll
  for (int r = 0; r < 8; ++r) {
#pragma unroll
    for (int off = 8; off >= 1; off >>= 1) {
      const float ov = __shfl_xor(bestVal[r], off, 32);
      const int   oi = __shfl_xor(bestIdx[r], off, 32);
      if (ov < bestVal[r] || (ov == bestVal[r] && oi < bestIdx[r])) {
        bestVal[r] = ov; bestIdx[r] = oi;
      }
    }
  }
  if (l16 == 0) {
#pragma unroll
    for (int r = 0; r < 8; ++r) {
      const int row = r + 8 * half;       // rows 0..7 in half 0, 8..15 in half 1
      const int Rg  = rowBase + row;
      idxLDS[wave * 16 + row] = bestIdx[r];
      out[Q_SIZE + 2 + Rg] = (float)bestIdx[r];   // encoding_indices (as float)
    }
  }
  __syncthreads();

  // ---- Quantized output (NCHW, coalesced) + squared-error loss ----
  const int   rowq  = l16;
  const int   Rq    = rowBase + rowq;
  const int   bq    = Rq >> 10;
  const int   hwq   = Rq & (HW - 1);
  const int   myIdx = idxLDS[wave * 16 + rowq];
  const float* erow = emb + (size_t)myIdx * EC;   // fp32 codebook row
  float lsum = 0.f;
  for (int c0 = 0; c0 < EC; c0 += 2) {
    const int c = c0 + half;                       // halves cover even/odd channels
    const size_t off = (((size_t)(bq * EC + c)) << 10) + hwq;
    const float q  = erow[c];
    const float xv = x[off];
    const float d  = q - xv;
    lsum += d * d;
    out[off] = q;                                  // q_st forward value == quantized
  }
  // Deterministic reduction: wave shuffle -> LDS -> fixed-order block sum.
#pragma unroll
  for (int off = 16; off >= 1; off >>= 1) lsum += __shfl_xor(lsum, off, 32);
  if (lane == 0) wpart[wave] = lsum;
  __syncthreads();
  if (tid == 0) {
    float s = 0.f;
#pragma unroll
    for (int w = 0; w < 8; ++w) s += wpart[w];
    wsPartial[blockIdx.x] = s;
  }
}

// Fixed-order final reduction: bit-stable across graph replays (no atomics).
__global__ void vq_finalize_kernel(const float* __restrict__ wsPartial,
                                   float* __restrict__ out) {
  float s = 0.f;
  for (int i = 0; i < NWG; ++i) s += wsPartial[i];
  s *= INV_TOTAL;
  out[Q_SIZE]     = s;   // dictionary_loss
  out[Q_SIZE + 1] = s;   // commitment_loss (same forward value)
}

extern "C" void kernel_launch(void* const* d_in, const int* in_sizes, int n_in,
                              void* d_out, int out_size, void* d_ws, size_t ws_size,
                              hipStream_t stream) {
  const float* x   = (const float*)d_in[0];   // [64,256,32,32] fp32
  const float* emb = (const float*)d_in[1];   // [512,256] fp32
  float* out = (float*)d_out;                 // q_st | dict | commit | indices
  float* ws  = (float*)d_ws;                  // 512 per-block loss partials

  vq_wmma_kernel<<<NWG, 256, SMEM_BYTES, stream>>>(x, emb, out, ws);
  vq_finalize_kernel<<<1, 1, 0, stream>>>(ws, out);
}
